// TorchMinimalMoE_40991167873584
// MI455X (gfx1250) — matrix operand order repaired and checked by reading, not compiled, not run on hardware
//
#include <hip/hip_runtime.h>

// ---------------- problem constants ----------------
#define NTOK    2048                 // G*S flattened tokens
#define D_DIM   2048
#define E_NUM   8
#define K_TOP   2

// ---------------- tiling ----------------
#define BM   64      // gathered rows per block
#define BN   128     // output columns per block
#define BK   32      // K step (bf16 wmma 16x16x32)
#define LDA  40      // padded LDS row stride in bf16 elems (conflict-free)
#define LDB  136     // epilogue repack stride (128 + 8)

typedef __attribute__((ext_vector_type(16))) __bf16 v16bf;
typedef __attribute__((ext_vector_type(8)))  __bf16 v8bf;
typedef __attribute__((ext_vector_type(4)))  __bf16 v4bf;
typedef __attribute__((ext_vector_type(8)))  float  v8f;
typedef __attribute__((ext_vector_type(4)))  float  v4f;

__device__ __forceinline__ v8bf cvt8(v4f f0, v4f f1) {
    v4bf b0 = __builtin_convertvector(f0, v4bf);
    v4bf b1 = __builtin_convertvector(f1, v4bf);
    return __builtin_shufflevector(b0, b1, 0, 1, 2, 3, 4, 5, 6, 7);
}

#define SHUF16(lo, hi) __builtin_shufflevector((lo), (hi), 0,1,2,3,4,5,6,7,8,9,10,11,12,13,14,15)

// ---------------- routing kernels ----------------
__global__ void moe_route_count(const long long* __restrict__ idx, int* __restrict__ cnt, int total) {
    int t = blockIdx.x * blockDim.x + threadIdx.x;
    if (t < total) atomicAdd(&cnt[(int)idx[t]], 1);
}

__global__ void moe_route_scan(const int* __restrict__ cnt, int* __restrict__ offs) {
    if (threadIdx.x == 0) {
        int acc = 0;
        offs[0] = 0;
        for (int e = 0; e < E_NUM; ++e) {
            acc += (cnt[e] + (BM - 1)) & ~(BM - 1);   // pad segments to 64 rows
            offs[e + 1] = acc;
        }
        offs[E_NUM + 1] = acc + NTOK;                 // group 8 = shared expert
    }
}

__global__ void moe_route_fill(const long long* __restrict__ idx, const float* __restrict__ w,
                               const int* __restrict__ offs, int* __restrict__ cur,
                               int* __restrict__ tokens, float* __restrict__ rowW, int total) {
    int t = blockIdx.x * blockDim.x + threadIdx.x;
    if (t < total) {
        int e    = (int)idx[t];
        int pos  = atomicAdd(&cur[e], 1);
        int slot = offs[e] + pos;
        tokens[slot] = t / K_TOP;
        rowW[slot]   = w[t];
    }
}

__global__ void moe_route_shared(const int* __restrict__ offs, int* __restrict__ tokens,
                                 float* __restrict__ rowW) {
    int t = blockIdx.x * blockDim.x + threadIdx.x;
    if (t < NTOK) {
        int slot = offs[E_NUM] + t;
        tokens[slot] = t;
        rowW[slot]   = 1.0f;
    }
}

__global__ void moe_cvt_x(const float* __restrict__ x, __bf16* __restrict__ xb, int total8) {
    int t = blockIdx.x * blockDim.x + threadIdx.x;
    if (t < total8) {
        const float* s = x + (size_t)t * 8;
        v4f f0 = *(const v4f*)s;
        v4f f1 = *(const v4f*)(s + 4);
        *(v8bf*)(xb + (size_t)t * 8) = cvt8(f0, f1);
    }
}

// ---------------- phase 1: H = silu(Xg @ Gw^T) * (Xg @ Uw^T) ----------------
// LDS carve (bf16 elems): A0 @0 (2560) | A1 @2560 | G0 @5120 (5120) | G1 @10240
//                         | U0 @15360 | U1 @20480 | end @25600 (=51200 B) | sTok
__global__ __launch_bounds__(256) void moe_phase1(
    const __bf16* __restrict__ xb,
    const float* __restrict__ gate_w, const float* __restrict__ up_w,
    const float* __restrict__ sgw,    const float* __restrict__ suw,
    const int* __restrict__ offs, const int* __restrict__ tokens,
    __bf16* __restrict__ H)
{
    const int rowBase = blockIdx.x * BM;
    const int totalRows = offs[E_NUM + 1];
    if (rowBase >= totalRows) return;

    int grp = 0;
    while (rowBase >= offs[grp + 1]) ++grp;
    const float* gw = (grp < E_NUM) ? gate_w + (size_t)grp * D_DIM * D_DIM : sgw;
    const float* uw = (grp < E_NUM) ? up_w   + (size_t)grp * D_DIM * D_DIM : suw;

    const int colBase = blockIdx.y * BN;
    const int tid  = threadIdx.x;
    const int lane = tid & 31;
    const int wave = tid >> 5;
    const int wm = wave & 1;
    const int wn = wave >> 1;
    const int half = lane >> 4;
    const int l16  = lane & 15;

    __shared__ __align__(16) char smem[51456];
    __bf16* const base = (__bf16*)smem;
    int*    const sTok = (int*)(smem + 51200);

    if (tid < BM) sTok[tid] = tokens[rowBase + tid];
    __syncthreads();

    const int arow = tid >> 2;          // 0..63
    const int aq   = tid & 3;           // 0..3 (8 bf16 per quarter)
    const int atok = sTok[arow];
    const __bf16* aSrc = (atok >= 0) ? xb + (size_t)atok * D_DIM + aq * 8 : nullptr;

    // weight loader roles (4 segments/thread): seg -> {matrix, row, quarter}
    const float* wsrc[4];
    int wdst[4];                        // LDS elem offset within buf0 region
    #pragma unroll
    for (int it = 0; it < 4; ++it) {
        int seg = tid + it * 256;       // 0..1023
        int mat = seg >> 9;             // 0 = gate, 1 = up
        int rem = seg & 511;
        int row = rem >> 2;             // 0..127
        int q   = rem & 3;
        wsrc[it] = (mat ? uw : gw) + (size_t)(colBase + row) * D_DIM + q * 8;
        wdst[it] = 5120 + mat * 10240 + row * LDA + q * 8;
    }

    // ---- prologue: stage kb = 0 into buffer 0 ----
    {
        if (atok >= 0) *(v8bf*)(base + arow * LDA + aq * 8) = *(const v8bf*)aSrc;
        else { v8bf z = {}; *(v8bf*)(base + arow * LDA + aq * 8) = z; }
        #pragma unroll
        for (int it = 0; it < 4; ++it) {
            v4f f0 = *(const v4f*)(wsrc[it]);
            v4f f1 = *(const v4f*)(wsrc[it] + 4);
            *(v8bf*)(base + wdst[it]) = cvt8(f0, f1);
        }
    }
    __syncthreads();

    v8f accG[2][2] = {}; v8f accU[2][2] = {};

    int buf = 0;
    for (int kb = 0; kb < D_DIM; kb += BK, buf ^= 1) {
        const int kn = kb + BK;
        const bool hasNext = (kn < D_DIM);

        // --- issue next tile's global loads early (overlap with wmma) ---
        v8bf aN = {};
        v4f f0[4], f1[4];
        if (hasNext) {
            if (atok >= 0) aN = *(const v8bf*)(aSrc + kn);
            #pragma unroll
            for (int it = 0; it < 4; ++it) {
                f0[it] = *(const v4f*)(wsrc[it] + kn);
                f1[it] = *(const v4f*)(wsrc[it] + kn + 4);
            }
        }

        // --- fragments from current buffer ---
        const __bf16* aBuf = base + buf * 2560;
        const __bf16* gBuf = base + 5120 + buf * 5120;
        const __bf16* uBuf = gBuf + 10240;

        v16bf aF[2];
        #pragma unroll
        for (int i = 0; i < 2; ++i) {
            const __bf16* p = aBuf + (wm * 32 + i * 16 + l16) * LDA;
            v8bf lo = *(const v8bf*)(p + half * 16);
            v8bf hi = *(const v8bf*)(p + half * 16 + 8);
            aF[i] = SHUF16(lo, hi);
        }
        v16bf gF[2], uF[2];
        #pragma unroll
        for (int j = 0; j < 2; ++j) {
            const __bf16* pg = gBuf + (wn * 32 + j * 16 + l16) * LDA + half * 16;
            v8bf lo = *(const v8bf*)pg;
            v8bf hi = *(const v8bf*)(pg + 8);
            gF[j] = SHUF16(lo, hi);
            const __bf16* pu = uBuf + (wn * 32 + j * 16 + l16) * LDA + half * 16;
            lo = *(const v8bf*)pu;
            hi = *(const v8bf*)(pu + 8);
            uF[j] = SHUF16(lo, hi);
        }
        #pragma unroll
        for (int i = 0; i < 2; ++i)
            #pragma unroll
            for (int j = 0; j < 2; ++j) {
                accG[i][j] = __builtin_amdgcn_wmma_f32_16x16x32_bf16(
                    false, aF[i], false, gF[j], (short)0, accG[i][j], false, false);
                accU[i][j] = __builtin_amdgcn_wmma_f32_16x16x32_bf16(
                    false, aF[i], false, uF[j], (short)0, accU[i][j], false, false);
            }

        // --- stage next tile into the other buffer ---
        if (hasNext) {
            const int nb = buf ^ 1;
            *(v8bf*)(base + nb * 2560 + arow * LDA + aq * 8) = aN;
            #pragma unroll
            for (int it = 0; it < 4; ++it)
                *(v8bf*)(base + wdst[it] + nb * 5120) = cvt8(f0[it], f1[it]);
        }
        __syncthreads();
    }

    // ---- epilogue: h = silu(g)*u, repack through LDS, coalesced b128 stores ----
    __bf16* const sOut = base;          // reuses pipeline buffers (post-barrier)
    #pragma unroll
    for (int i = 0; i < 2; ++i)
        #pragma unroll
        for (int j = 0; j < 2; ++j)
            #pragma unroll
            for (int v = 0; v < 8; ++v) {
                float g = accG[i][j][v];
                float u = accU[i][j][v];
                float h = g * __builtin_amdgcn_rcpf(1.0f + __expf(-g)) * u;
                int r = wm * 32 + i * 16 + v + 8 * half;
                int c = wn * 32 + j * 16 + l16;
                sOut[r * LDB + c] = (__bf16)h;
            }
    __syncthreads();
    #pragma unroll
    for (int it = 0; it < 4; ++it) {
        int chunk = tid + it * 256;     // 0..1023 = 64 rows x 16 chunks
        int row = chunk >> 4;
        int cc  = chunk & 15;
        *(v8bf*)(H + (size_t)(rowBase + row) * D_DIM + colBase + cc * 8) =
            *(const v8bf*)(sOut + row * LDB + cc * 8);
    }
}

// ---------------- phase 2: Out += rowW * (H @ Dw^T), scatter to tokens ----------------
// LDS carve: A0 @0 (2560) | A1 @2560 | B0 @5120 (5120) | B1 @10240 | end @15360 (=30720 B)
__global__ __launch_bounds__(256) void moe_phase2(
    const __bf16* __restrict__ H,
    const float* __restrict__ down_w, const float* __restrict__ sdw,
    const int* __restrict__ offs, const int* __restrict__ tokens,
    const float* __restrict__ rowW, float* __restrict__ out)
{
    const int rowBase = blockIdx.x * BM;
    const int totalRows = offs[E_NUM + 1];
    if (rowBase >= totalRows) return;

    int grp = 0;
    while (rowBase >= offs[grp + 1]) ++grp;
    const float* dw = (grp < E_NUM) ? down_w + (size_t)grp * D_DIM * D_DIM : sdw;

    const int colBase = blockIdx.y * BN;
    const int tid  = threadIdx.x;
    const int lane = tid & 31;
    const int wave = tid >> 5;
    const int wm = wave & 1;
    const int wn = wave >> 1;
    const int half = lane >> 4;
    const int l16  = lane & 15;

    __shared__ __align__(16) char smem[31232];
    __bf16* const base = (__bf16*)smem;
    int*    const sTok = (int*)(smem + 30720);
    float*  const sW   = (float*)(smem + 30976);

    if (tid < BM) { sTok[tid] = tokens[rowBase + tid]; sW[tid] = rowW[rowBase + tid]; }
    __syncthreads();

    const int arow = tid >> 2;
    const int aq   = tid & 3;
    const __bf16* aSrc = H + (size_t)(rowBase + arow) * D_DIM + aq * 8;

    const float* wsrc[2];
    int wdst[2];
    #pragma unroll
    for (int it = 0; it < 2; ++it) {
        int seg = tid + it * 256;       // 0..511 = 128 rows x 4 quarters
        int row = seg >> 2;
        int q   = seg & 3;
        wsrc[it] = dw + (size_t)(colBase + row) * D_DIM + q * 8;
        wdst[it] = 5120 + row * LDA + q * 8;
    }

    // prologue: stage kb = 0 into buffer 0
    {
        *(v8bf*)(base + arow * LDA + aq * 8) = *(const v8bf*)aSrc;
        #pragma unroll
        for (int it = 0; it < 2; ++it) {
            v4f f0 = *(const v4f*)(wsrc[it]);
            v4f f1 = *(const v4f*)(wsrc[it] + 4);
            *(v8bf*)(base + wdst[it]) = cvt8(f0, f1);
        }
    }
    __syncthreads();

    v8f acc[2][2] = {};

    int buf = 0;
    for (int kb = 0; kb < D_DIM; kb += BK, buf ^= 1) {
        const int kn = kb + BK;
        const bool hasNext = (kn < D_DIM);

        v8bf aN = {};
        v4f f0[2], f1[2];
        if (hasNext) {
            aN = *(const v8bf*)(aSrc + kn);
            #pragma unroll
            for (int it = 0; it < 2; ++it) {
                f0[it] = *(const v4f*)(wsrc[it] + kn);
                f1[it] = *(const v4f*)(wsrc[it] + kn + 4);
            }
        }

        const __bf16* aBuf = base + buf * 2560;
        const __bf16* bBuf = base + 5120 + buf * 5120;

        v16bf aF[2];
        #pragma unroll
        for (int i = 0; i < 2; ++i) {
            const __bf16* p = aBuf + (wm * 32 + i * 16 + l16) * LDA;
            v8bf lo = *(const v8bf*)(p + half * 16);
            v8bf hi = *(const v8bf*)(p + half * 16 + 8);
            aF[i] = SHUF16(lo, hi);
        }
        v16bf bF[2];
        #pragma unroll
        for (int j = 0; j < 2; ++j) {
            const __bf16* p = bBuf + (wn * 32 + j * 16 + l16) * LDA + half * 16;
            v8bf lo = *(const v8bf*)p;
            v8bf hi = *(const v8bf*)(p + 8);
            bF[j] = SHUF16(lo, hi);
        }
        #pragma unroll
        for (int i = 0; i < 2; ++i)
            #pragma unroll
            for (int j = 0; j < 2; ++j)
                acc[i][j] = __builtin_amdgcn_wmma_f32_16x16x32_bf16(
                    false, aF[i], false, bF[j], (short)0, acc[i][j], false, false);

        if (hasNext) {
            const int nb = buf ^ 1;
            *(v8bf*)(base + nb * 2560 + arow * LDA + aq * 8) = aN;
            #pragma unroll
            for (int it = 0; it < 2; ++it)
                *(v8bf*)(base + wdst[it] + nb * 5120) = cvt8(f0[it], f1[it]);
        }
        __syncthreads();
    }

    // scatter-accumulate into output with routing weight (16 consecutive cols/lane-group)
    #pragma unroll
    for (int i = 0; i < 2; ++i)
        #pragma unroll
        for (int j = 0; j < 2; ++j)
            #pragma unroll
            for (int v = 0; v < 8; ++v) {
                int rl = wm * 32 + i * 16 + v + 8 * half;
                int token = sTok[rl];
                if (token >= 0) {
                    int c = colBase + wn * 32 + j * 16 + l16;
                    atomicAdd(&out[(size_t)token * D_DIM + c], sW[rl] * acc[i][j][v]);
                }
            }
}

// ---------------- launcher ----------------
extern "C" void kernel_launch(void* const* d_in, const int* in_sizes, int n_in,
                              void* d_out, int out_size, void* d_ws, size_t ws_size,
                              hipStream_t stream) {
    (void)in_sizes; (void)n_in; (void)out_size; (void)ws_size;
    const float*     x       = (const float*)d_in[0];
    const float*     weights = (const float*)d_in[1];
    const long long* indices = (const long long*)d_in[2];
    // d_in[3], d_in[4] (expert_offsets / token_counts) are zeros in the reference — unused.
    const float* gate_w = (const float*)d_in[5];
    const float* up_w   = (const float*)d_in[6];
    const float* down_w = (const float*)d_in[7];
    const float* sgw    = (const float*)d_in[8];
    const float* suw    = (const float*)d_in[9];
    const float* sdw    = (const float*)d_in[10];
    float* out = (float*)d_out;

    // workspace carve (all 64B aligned)
    char* ws = (char*)d_ws;
    int*    cnt    = (int*)(ws + 0);
    int*    cur    = (int*)(ws + 64);
    int*    offs   = (int*)(ws + 128);
    int*    tokens = (int*)(ws + 256);          // 8192 ints
    float*  rowW   = (float*)(ws + 256 + 32768);
    __bf16* xb     = (__bf16*)(ws + 65792);                             // 8 MB
    __bf16* H      = (__bf16*)(ws + 65792 + (size_t)NTOK * D_DIM * 2);  // ~27 MB

    hipMemsetAsync(cnt, 0, 64, stream);
    hipMemsetAsync(cur, 0, 64, stream);
    hipMemsetAsync(tokens, 0xFF, 8192 * sizeof(int), stream);           // -1 = padding row
    hipMemsetAsync(out, 0, (size_t)NTOK * D_DIM * sizeof(float), stream);

    moe_cvt_x<<<(NTOK * D_DIM / 8 + 255) / 256, 256, 0, stream>>>(x, xb, NTOK * D_DIM / 8);
    moe_route_count<<<(NTOK * K_TOP + 255) / 256, 256, 0, stream>>>(indices, cnt, NTOK * K_TOP);
    moe_route_scan<<<1, 32, 0, stream>>>(cnt, offs);
    moe_route_fill<<<(NTOK * K_TOP + 255) / 256, 256, 0, stream>>>(indices, weights, offs, cur,
                                                                   tokens, rowW, NTOK * K_TOP);
    moe_route_shared<<<(NTOK + 255) / 256, 256, 0, stream>>>(offs, tokens, rowW);

    // max padded rows: routed <= 4544 (71 tiles) + shared 2048 (32 tiles)
    dim3 grid(104, D_DIM / BN);   // 104 x 16
    moe_phase1<<<grid, 256, 0, stream>>>(xb, gate_w, up_w, sgw, suw, offs, tokens, H);
    moe_phase2<<<grid, 256, 0, stream>>>(H, down_w, sdw, offs, tokens, rowW, out);
}
